// MoELayer_34703335752416
// MI455X (gfx1250) — matrix operand order repaired and checked by reading, not compiled, not run on hardware
//
#include <hip/hip_runtime.h>
#include <hip/hip_bf16.h>

// ---------------- problem constants ----------------
#define HDIM 1024
#define FDIM 4096
#define NEXP 8
#define KSEL 2
#define NTOK 2048              // B*S = 2*1024
#define ROWS_TOTAL (NTOK*KSEL) // 4096 token-expert rows, always
#define LDSK 40                // padded LDS row (elements): 80B, 16B-aligned frag reads

typedef __bf16 bf16_t;
typedef __attribute__((ext_vector_type(16))) __bf16 v16bf;
typedef __attribute__((ext_vector_type(8)))  __bf16 v8bf;
typedef __attribute__((ext_vector_type(8)))  float  v8f;

union FragBF {
    v16bf v;
    v8bf  h[2];
};

union Pack4 {                 // 4 bf16 packed for one ds_store_b64
    bf16_t h[4];
    unsigned long long u64;
};

// ---------------- optional CDNA5 async copy to LDS ----------------
#if defined(__has_builtin)
#if __has_builtin(__builtin_amdgcn_global_load_async_to_lds_b32) && \
    __has_builtin(__builtin_amdgcn_s_wait_asynccnt)
#define USE_ASYNC_LDS 1
#endif
#endif

__device__ __forceinline__ void ldscpy_b32(void* lds, const void* g) {
#ifdef USE_ASYNC_LDS
    __builtin_amdgcn_global_load_async_to_lds_b32(
        (__attribute__((address_space(1))) int*)(g),
        (__attribute__((address_space(3))) int*)(lds), 0, 0);
#else
    *(unsigned int*)lds = *(const unsigned int*)g;
#endif
}

__device__ __forceinline__ void async_fence() {
#ifdef USE_ASYNC_LDS
    __builtin_amdgcn_s_wait_asynccnt(0);
#endif
}

// ---------------- workspace layout (bytes) ----------------
#define OFF_XB      ((size_t)0)
#define OFF_HMID    (OFF_XB + (size_t)NTOK*HDIM*2)
#define OFF_TOKIDX  (OFF_HMID + (size_t)ROWS_TOTAL*FDIM*2)
#define OFF_TOKW    (OFF_TOKIDX + (size_t)NEXP*NTOK*4)
#define OFF_COUNTS  (OFF_TOKW + (size_t)NEXP*NTOK*4)
#define OFF_ROWOFF  (OFF_COUNTS + (size_t)64)

// ---------------- kernel 1: zero out + counts ----------------
__global__ void moe_zero_kernel(float* out, int n_out, int* counts) {
    int i = blockIdx.x * blockDim.x + threadIdx.x;
    for (; i < n_out; i += gridDim.x * blockDim.x) out[i] = 0.0f;
    if (blockIdx.x == 0 && threadIdx.x < NEXP) counts[threadIdx.x] = 0;
}

// ---------------- kernel 2: router (one wave32 per token) ----------------
__global__ void moe_router_kernel(const float* __restrict__ x,
                                  const float* __restrict__ Wr,
                                  int* counts, int* tokidx, float* tokw) {
    const int lane = threadIdx.x & 31;
    const int wv   = threadIdx.x >> 5;
    const int t    = blockIdx.x * 8 + wv;
    if (t >= NTOK) return;

    float acc[NEXP];
#pragma unroll
    for (int e = 0; e < NEXP; ++e) acc[e] = 0.0f;

    const float* xrow = x + (size_t)t * HDIM;
    for (int j = lane; j < HDIM; j += 32) {
        float xv = xrow[j];
#pragma unroll
        for (int e = 0; e < NEXP; ++e) acc[e] += xv * Wr[e * HDIM + j];
    }
#pragma unroll
    for (int e = 0; e < NEXP; ++e) {
#pragma unroll
        for (int off = 16; off > 0; off >>= 1)
            acc[e] += __shfl_down(acc[e], off, 32);
    }
    if (lane == 0) {
        float m = acc[0];
#pragma unroll
        for (int e = 1; e < NEXP; ++e) m = fmaxf(m, acc[e]);
        float p[NEXP];
#pragma unroll
        for (int e = 0; e < NEXP; ++e) p[e] = __expf(acc[e] - m);
        int e1 = 0;
#pragma unroll
        for (int e = 1; e < NEXP; ++e) if (p[e] > p[e1]) e1 = e;
        int e2 = (e1 == 0) ? 1 : 0;
#pragma unroll
        for (int e = 0; e < NEXP; ++e) if (e != e1 && p[e] > p[e2]) e2 = e;
        float s  = p[e1] + p[e2];
        int s1 = atomicAdd(&counts[e1], 1);
        tokidx[e1 * NTOK + s1] = t;
        tokw  [e1 * NTOK + s1] = p[e1] / s;
        int s2 = atomicAdd(&counts[e2], 1);
        tokidx[e2 * NTOK + s2] = t;
        tokw  [e2 * NTOK + s2] = p[e2] / s;
    }
}

// ---------------- kernel 3: x fp32 -> bf16 (vectorized) ----------------
__global__ void moe_xconv_kernel(const float* __restrict__ x,
                                 bf16_t* __restrict__ xb, int n4) {
    int i = blockIdx.x * blockDim.x + threadIdx.x;
    for (; i < n4; i += gridDim.x * blockDim.x) {
        float4 v = ((const float4*)x)[i];
        Pack4 p;
        p.h[0] = (bf16_t)v.x; p.h[1] = (bf16_t)v.y;
        p.h[2] = (bf16_t)v.z; p.h[3] = (bf16_t)v.w;
        ((unsigned long long*)xb)[i] = p.u64;
    }
}

// ---------------- kernel 4: prefix sum of counts ----------------
__global__ void moe_prefix_kernel(const int* counts, int* rowoff) {
    if (threadIdx.x == 0 && blockIdx.x == 0) {
        int off = 0;
        for (int e = 0; e < NEXP; ++e) { rowoff[e] = off; off += counts[e]; }
    }
}

// ---------------- kernel 5: grouped gate/up GEMM + SwiGLU ----------------
// grid: (FDIM/128, NTOK/16, NEXP), block: 256 (8 waves, 16 N-cols each)
__global__ void __launch_bounds__(256)
moe_gateup_kernel(const bf16_t* __restrict__ xb,
                  const float* __restrict__ W1,
                  const float* __restrict__ W3,
                  const int* __restrict__ tokidx,
                  const int* __restrict__ counts,
                  const int* __restrict__ rowoff,
                  bf16_t* __restrict__ hmid) {
    const int e     = blockIdx.z;
    const int cnt   = counts[e];
    const int mtile = blockIdx.y;
    if (mtile * 16 >= cnt) return;

    const int tid  = threadIdx.x;
    const int lane = tid & 31;
    const int wv   = tid >> 5;
    const int n0   = blockIdx.x * 128;

    __shared__ __align__(16) bf16_t sA [2][16][LDSK];
    __shared__ __align__(16) bf16_t sW1[2][128][LDSK];
    __shared__ __align__(16) bf16_t sW3[2][128][LDSK];

    // ---- hoisted per-thread A-gather mapping (loop-invariant) ----
    const int am  = tid >> 4;          // 0..15, one LDS row per thread
    const int ak  = (tid & 15) * 2;    // 2 bf16 (4 bytes) per thread per step
    const int agm = mtile * 16 + am;
    const bool avalid = (agm < cnt);
    const bf16_t* arow =
        xb + (size_t)(avalid ? tokidx[e * NTOK + agm] : 0) * HDIM;
    if (!avalid) {                      // pre-zero invalid rows once (both buffers)
        sA[0][am][ak] = (bf16_t)0.0f; sA[0][am][ak + 1] = (bf16_t)0.0f;
        sA[1][am][ak] = (bf16_t)0.0f; sA[1][am][ak + 1] = (bf16_t)0.0f;
    }

    const size_t Wbase = (size_t)e * HDIM * FDIM;

    // ---- per-thread 4x4 W staging sub-block ----
    const int snn = (tid & 31) * 4;    // 4 consecutive N columns
    const int skb = (tid >> 5) * 4;    // 4 consecutive K rows

    // ---- fragment mappings (§7.12.2) ----
    const int mrow = lane & 15;
    const int kb   = (lane >> 4) * 16;
    const int nloc = wv * 16 + (lane & 15);
    const int bk0  = (lane >> 4) * 16;

    v8f accG = {}; v8f accU = {};

    auto stage = [&](int kt, int b) {
        const int k0 = kt * 32;
        if (avalid) ldscpy_b32(&sA[b][am][ak], arow + k0 + ak);
        float r1[4][4], r3[4][4];
#pragma unroll
        for (int r = 0; r < 4; ++r) {   // coalesced float4 row loads
            const size_t gidx = Wbase + (size_t)(k0 + skb + r) * FDIM + (n0 + snn);
            float4 w1 = *(const float4*)&W1[gidx];
            float4 w3 = *(const float4*)&W3[gidx];
            r1[r][0] = w1.x; r1[r][1] = w1.y; r1[r][2] = w1.z; r1[r][3] = w1.w;
            r3[r][0] = w3.x; r3[r][1] = w3.y; r3[r][2] = w3.z; r3[r][3] = w3.w;
        }
#pragma unroll
        for (int q = 0; q < 4; ++q) {   // register transpose -> packed b64 stores
            Pack4 p1, p3;
#pragma unroll
            for (int r = 0; r < 4; ++r) {
                p1.h[r] = (bf16_t)r1[r][q];
                p3.h[r] = (bf16_t)r3[r][q];
            }
            *(unsigned long long*)&sW1[b][snn + q][skb] = p1.u64;
            *(unsigned long long*)&sW3[b][snn + q][skb] = p3.u64;
        }
    };

    stage(0, 0);                        // prologue
    const int KT = HDIM / 32;
    for (int kt = 0; kt < KT; ++kt) {
        const int buf = kt & 1;
        async_fence();
        __syncthreads();                // tile 'kt' staging complete
        if (kt + 1 < KT) stage(kt + 1, buf ^ 1);   // overlap with WMMA below
        if (kt + 2 < KT)                            // -> global_prefetch_b8
            __builtin_prefetch(&W1[Wbase + (size_t)((kt + 2) * 32) * FDIM + n0 + (tid & 127)], 0, 1);

        FragBF a, b1, b3;
        a.h[0]  = *(const v8bf*)&sA[buf][mrow][kb];
        a.h[1]  = *(const v8bf*)&sA[buf][mrow][kb + 8];
        b1.h[0] = *(const v8bf*)&sW1[buf][nloc][bk0];
        b1.h[1] = *(const v8bf*)&sW1[buf][nloc][bk0 + 8];
        b3.h[0] = *(const v8bf*)&sW3[buf][nloc][bk0];
        b3.h[1] = *(const v8bf*)&sW3[buf][nloc][bk0 + 8];

        accG = __builtin_amdgcn_wmma_f32_16x16x32_bf16(false, a.v, false, b1.v,
                                                       (short)0, accG, false, false);
        accU = __builtin_amdgcn_wmma_f32_16x16x32_bf16(false, a.v, false, b3.v,
                                                       (short)0, accU, false, false);
    }

    // epilogue: h = silu(gate) * up -> bf16 compact hmid
    const int mbase = (lane >> 4) * 8;
    const int col   = n0 + nloc;
    const int rbase = rowoff[e];
#pragma unroll
    for (int r = 0; r < 8; ++r) {
        int gm = mtile * 16 + mbase + r;
        if (gm < cnt) {
            float g = accG[r], u = accU[r];
            float hv = (g / (1.0f + __expf(-g))) * u;
            hmid[(size_t)(rbase + gm) * FDIM + col] = (bf16_t)hv;
        }
    }
}

// ---------------- kernel 6: grouped down GEMM, scaled accumulate ----------------
// grid: (HDIM/128, NTOK/16, NEXP), block: 256
__global__ void __launch_bounds__(256)
moe_down_kernel(const bf16_t* __restrict__ hmid,
                const float* __restrict__ W2,
                const int* __restrict__ tokidx,
                const float* __restrict__ tokw,
                const int* __restrict__ counts,
                const int* __restrict__ rowoff,
                float* __restrict__ out) {
    const int e     = blockIdx.z;
    const int cnt   = counts[e];
    const int mtile = blockIdx.y;
    if (mtile * 16 >= cnt) return;

    const int tid  = threadIdx.x;
    const int lane = tid & 31;
    const int wv   = tid >> 5;
    const int n0   = blockIdx.x * 128;

    __shared__ __align__(16) bf16_t sA [2][16][LDSK];
    __shared__ __align__(16) bf16_t sW2[2][128][LDSK];

    const int am  = tid >> 4;
    const int ak  = (tid & 15) * 2;
    const int agm = mtile * 16 + am;
    const bool avalid = (agm < cnt);
    const int rbase = rowoff[e];
    const bf16_t* arow = hmid + (size_t)(rbase + (avalid ? agm : 0)) * FDIM;
    if (!avalid) {
        sA[0][am][ak] = (bf16_t)0.0f; sA[0][am][ak + 1] = (bf16_t)0.0f;
        sA[1][am][ak] = (bf16_t)0.0f; sA[1][am][ak + 1] = (bf16_t)0.0f;
    }

    const size_t Wbase = (size_t)e * FDIM * HDIM;   // W2: [E][F][H]

    const int snn = (tid & 31) * 4;
    const int skb = (tid >> 5) * 4;

    const int mrow = lane & 15;
    const int kb   = (lane >> 4) * 16;
    const int nloc = wv * 16 + (lane & 15);
    const int bk0  = (lane >> 4) * 16;

    v8f acc = {};

    auto stage = [&](int kt, int b) {
        const int k0 = kt * 32;
        if (avalid) ldscpy_b32(&sA[b][am][ak], arow + k0 + ak);
        float rw[4][4];
#pragma unroll
        for (int r = 0; r < 4; ++r) {
            float4 w = *(const float4*)&W2[Wbase + (size_t)(k0 + skb + r) * HDIM + (n0 + snn)];
            rw[r][0] = w.x; rw[r][1] = w.y; rw[r][2] = w.z; rw[r][3] = w.w;
        }
#pragma unroll
        for (int q = 0; q < 4; ++q) {
            Pack4 p;
#pragma unroll
            for (int r = 0; r < 4; ++r) p.h[r] = (bf16_t)rw[r][q];
            *(unsigned long long*)&sW2[b][snn + q][skb] = p.u64;
        }
    };

    stage(0, 0);
    const int KT = FDIM / 32;
    for (int kt = 0; kt < KT; ++kt) {
        const int buf = kt & 1;
        async_fence();
        __syncthreads();
        if (kt + 1 < KT) stage(kt + 1, buf ^ 1);
        if (kt + 2 < KT)
            __builtin_prefetch(&W2[Wbase + (size_t)((kt + 2) * 32) * HDIM + n0 + (tid & 127)], 0, 1);

        FragBF a, b;
        a.h[0] = *(const v8bf*)&sA[buf][mrow][kb];
        a.h[1] = *(const v8bf*)&sA[buf][mrow][kb + 8];
        b.h[0] = *(const v8bf*)&sW2[buf][nloc][bk0];
        b.h[1] = *(const v8bf*)&sW2[buf][nloc][bk0 + 8];

        acc = __builtin_amdgcn_wmma_f32_16x16x32_bf16(false, a.v, false, b.v,
                                                      (short)0, acc, false, false);
    }

    const int mbase = (lane >> 4) * 8;
    const int col   = n0 + nloc;
#pragma unroll
    for (int r = 0; r < 8; ++r) {
        int gm = mtile * 16 + mbase + r;
        if (gm < cnt) {
            int   t  = tokidx[e * NTOK + gm];
            float wt = tokw [e * NTOK + gm];
            atomicAdd(&out[(size_t)t * HDIM + col], wt * acc[r]);
        }
    }
}

// ---------------- launcher ----------------
extern "C" void kernel_launch(void* const* d_in, const int* in_sizes, int n_in,
                              void* d_out, int out_size, void* d_ws, size_t ws_size,
                              hipStream_t stream) {
    const float* x  = (const float*)d_in[0];   // [B,S,H]
    const float* Wr = (const float*)d_in[1];   // [E,H]
    const float* W1 = (const float*)d_in[2];   // [E,H,F]
    const float* W2 = (const float*)d_in[3];   // [E,F,H]
    const float* W3 = (const float*)d_in[4];   // [E,H,F]
    float* out = (float*)d_out;                // [B,S,H] fp32

    char* ws = (char*)d_ws;
    bf16_t* xb     = (bf16_t*)(ws + OFF_XB);
    bf16_t* hmid   = (bf16_t*)(ws + OFF_HMID);
    int*    tokidx = (int*)   (ws + OFF_TOKIDX);
    float*  tokw   = (float*) (ws + OFF_TOKW);
    int*    counts = (int*)   (ws + OFF_COUNTS);
    int*    rowoff = (int*)   (ws + OFF_ROWOFF);

    moe_zero_kernel<<<1024, 256, 0, stream>>>(out, NTOK * HDIM, counts);
    moe_router_kernel<<<NTOK / 8, 256, 0, stream>>>(x, Wr, counts, tokidx, tokw);
    moe_xconv_kernel<<<1024, 256, 0, stream>>>(x, xb, (NTOK * HDIM) / 4);
    moe_prefix_kernel<<<1, 1, 0, stream>>>(counts, rowoff);

    dim3 gGU(FDIM / 128, NTOK / 16, NEXP);
    moe_gateup_kernel<<<gGU, 256, 0, stream>>>(xb, W1, W3, tokidx, counts, rowoff, hmid);

    dim3 gDN(HDIM / 128, NTOK / 16, NEXP);
    moe_down_kernel<<<gDN, 256, 0, stream>>>(hmid, W2, tokidx, tokw, counts, rowoff, out);
}
